// GNN_Generator_60430189855244
// MI455X (gfx1250) — matrix operand order repaired and checked by reading, not compiled, not run on hardware
//
#include <hip/hip_runtime.h>
#include <hip/hip_bf16.h>
#include <math.h>

// ---------------------------------------------------------------------------
// CDNA5 (gfx1250) WMMA implementation of the GNN generator.
// All matmuls run on v_wmma_f32_16x16x32_f16 (f16 in, f32 accumulate).
// Weights are pre-packed into B-fragment order so each lane loads its 32-byte
// v16h fragment with one vector load (L2-resident, ~256KB total).
// ---------------------------------------------------------------------------

typedef __attribute__((ext_vector_type(16))) _Float16 v16h;
typedef __attribute__((ext_vector_type(8)))  float    v8f;

#define LDH 136   // LDS activation row pitch (halves); 272B rows = 16B aligned

// ---- A fragment from global f32 matrix (16 rows x 32 K-slice) --------------
// Layout (ISA 7.12.2, 16-bit A 16x32): lanes 0-15 rows 0-15 hold K=k0+[0..7],
// halves 8..15 hold K=k0+[16..23]; lanes 16-31 hold K=k0+[8..15] / [24..31].
__device__ __forceinline__ v16h load_A_g(const float* __restrict__ x, int ld,
                                         int m0, int k0, int lane) {
  const int m  = m0 + (lane & 15);
  const int kb = k0 + ((lane >> 4) << 3);
  const float* p = x + (size_t)m * ld + kb;
  v16h a;
#pragma unroll
  for (int i = 0; i < 8; ++i) {
    a[i]     = (_Float16)p[i];
    a[i + 8] = (_Float16)p[i + 16];
  }
  return a;
}

// ---- A fragment from wave-private LDS f16 tile ------------------------------
__device__ __forceinline__ v16h load_A_lds(const _Float16* Hw, int k0, int lane) {
  const int m  = lane & 15;
  const int kb = k0 + ((lane >> 4) << 3);
  const _Float16* p = Hw + m * LDH + kb;
  v16h a;
#pragma unroll
  for (int i = 0; i < 8; ++i) {   // two 16B contiguous chunks -> ds_load_b128 x2
    a[i]     = p[i];
    a[i + 8] = p[i + 16];
  }
  return a;
}

// ---- A fragment broadcasting z (same 32 values for every row) ---------------
__device__ __forceinline__ v16h load_A_zbc(const float* __restrict__ z, int lane) {
  const int kb = (lane >> 4) << 3;
  v16h a;
#pragma unroll
  for (int i = 0; i < 8; ++i) {
    a[i]     = (_Float16)z[kb + i];
    a[i + 8] = (_Float16)z[kb + 16 + i];
  }
  return a;
}

// ---- one row of WMMAs over NTILES 16-wide output tiles ----------------------
template<int NTILES>
__device__ __forceinline__ void wmma_row(const v16h a, const _Float16* __restrict__ Wp,
                                         int kc, v8f* acc, int lane) {
#pragma unroll
  for (int nt = 0; nt < NTILES; ++nt) {
    const v16h b = *(const v16h*)(Wp + ((((size_t)kc * NTILES + nt) * 32 + lane) << 4));
    acc[nt] = __builtin_amdgcn_wmma_f32_16x16x32_f16(false, a, false, b,
                                                     (short)0, acc[nt], false, false);
  }
}

template<int KTILES, int NTILES>
__device__ __forceinline__ void gemm_gA(const float* __restrict__ x, int ld, int m0,
                                        const _Float16* __restrict__ Wp, v8f* acc, int lane) {
#pragma unroll
  for (int kc = 0; kc < KTILES; ++kc) {
    const v16h a = load_A_g(x, ld, m0, kc * 32, lane);
    wmma_row<NTILES>(a, Wp, kc, acc, lane);
  }
}

template<int KTILES, int NTILES>
__device__ __forceinline__ void gemm_lA(const _Float16* Hw,
                                        const _Float16* __restrict__ Wp, v8f* acc, int lane) {
#pragma unroll
  for (int kc = 0; kc < KTILES; ++kc) {
    const v16h a = load_A_lds(Hw, kc * 32, lane);
    wmma_row<NTILES>(a, Wp, kc, acc, lane);
  }
}

// ---- bias (+ReLU) and f16 store of C tiles into the wave's LDS tile ---------
// C layout: lane holds col n0+(lane&15); VGPR r holds row r + (lane>=16 ? 8:0).
template<int NTILES, bool RELU>
__device__ __forceinline__ void store_act_lds(const v8f* acc, const float* __restrict__ bias,
                                              _Float16* Hw, int lane) {
  const int nl = lane & 15, mb = (lane >> 4) << 3;
#pragma unroll
  for (int nt = 0; nt < NTILES; ++nt) {
    const float bv = bias[nt * 16 + nl];
#pragma unroll
    for (int r = 0; r < 8; ++r) {
      float v = acc[nt][r] + bv;
      if (RELU) v = fmaxf(v, 0.0f);
      Hw[(mb + r) * LDH + nt * 16 + nl] = (_Float16)v;
    }
  }
}

// ---------------------------------------------------------------------------
// Weight pre-pack: f32 KxN row-major -> f16 fragments in register-load order.
// Fragment element (kc, nt, lane, h) <- W[(kc*32 + (lane>=16?8:0) + (h&7) + (h>=8?16:0)) * N + nt*16 + lane%16]
// ---------------------------------------------------------------------------
__global__ void k_pack(const float* __restrict__ W, _Float16* __restrict__ dst,
                       int K, int N) {
  const int t = blockIdx.x * 256 + threadIdx.x;
  if (t >= K * N) return;
  const int h      = t & 15;
  const int lane   = (t >> 4) & 31;
  const int rem    = t >> 9;
  const int ntiles = N >> 4;
  const int nt = rem % ntiles;
  const int kc = rem / ntiles;
  const int n = nt * 16 + (lane & 15);
  const int k = kc * 32 + ((lane >> 4) << 3) + (h & 7) + ((h >> 3) << 4);
  dst[t] = (_Float16)W[(size_t)k * N + n];
}

// Deterministic fixed-order reduction of per-block partial column sums.
__global__ void k_reduce(const float* __restrict__ partial, float* __restrict__ s,
                         int nblocks) {
  const int c = threadIdx.x;  // 64 threads
  float t = 0.0f;
  for (int b = 0; b < nblocks; ++b) t += partial[(size_t)b * 64 + c];
  s[c] = t;
}

// ---------------------------------------------------------------------------
// Kernel 1: inn = in_MLP(x) (64->128->128->64), write inn and per-block
// column partial sums (for s = sum_j inn_j).
// ---------------------------------------------------------------------------
__global__ void __launch_bounds__(256) k_msg_in(
    const float* __restrict__ x,
    const _Float16* __restrict__ W1, const float* __restrict__ b1,
    const _Float16* __restrict__ W2, const float* __restrict__ b2,
    const _Float16* __restrict__ W3, const float* __restrict__ b3,
    float* __restrict__ inn, float* __restrict__ partial) {
  __shared__ _Float16 H[8 * 16 * LDH];
  __shared__ float wsum[8][64];
  const int tid = threadIdx.x, wave = tid >> 5, lane = tid & 31;
  _Float16* Hw = H + wave * (16 * LDH);
  const int node0 = blockIdx.x * 128 + wave * 16;
  const v8f vzero = {0.f, 0.f, 0.f, 0.f, 0.f, 0.f, 0.f, 0.f};

  v8f acc[8];
#pragma unroll
  for (int i = 0; i < 8; ++i) acc[i] = vzero;
  gemm_gA<2, 8>(x, 64, node0, W1, acc, lane);          // 64 -> 128
  store_act_lds<8, true>(acc, b1, Hw, lane);
#pragma unroll
  for (int i = 0; i < 8; ++i) acc[i] = vzero;
  gemm_lA<4, 8>(Hw, W2, acc, lane);                    // 128 -> 128
  store_act_lds<8, true>(acc, b2, Hw, lane);
  v8f acc3[4];
#pragma unroll
  for (int i = 0; i < 4; ++i) acc3[i] = vzero;
  gemm_lA<4, 4>(Hw, W3, acc3, lane);                   // 128 -> 64 (no relu)

  const int nl = lane & 15, mb = (lane >> 4) << 3;
#pragma unroll
  for (int nt = 0; nt < 4; ++nt) {
    const float bv = b3[nt * 16 + nl];
    float part = 0.0f;
#pragma unroll
    for (int r = 0; r < 8; ++r) {
      const float v = acc3[nt][r] + bv;
      inn[(size_t)(node0 + mb + r) * 64 + nt * 16 + nl] = v;
      part += v;
    }
    part += __shfl_xor(part, 16, 32);  // combine the two 8-row lane halves
    if (lane < 16) wsum[wave][nt * 16 + nl] = part;
  }
  __syncthreads();
  if (tid < 64) {
    float t = 0.0f;
#pragma unroll
    for (int w = 0; w < 8; ++w) t += wsum[w][tid];     // fixed order -> deterministic
    partial[(size_t)blockIdx.x * 64 + tid] = t;
  }
}

// ---------------------------------------------------------------------------
// Kernel 2: avg = (cur_MLP(x) + s - inn) / N ; xout = fin_MLP(avg).
// Safe in place: row i only reads row i of x plus the already-final inn/s.
// ---------------------------------------------------------------------------
__global__ void __launch_bounds__(256) k_msg_fin(
    const float* __restrict__ x, const float* __restrict__ inn,
    const float* __restrict__ s,
    const _Float16* __restrict__ cW1, const float* __restrict__ cb1,
    const _Float16* __restrict__ cW2, const float* __restrict__ cb2,
    const _Float16* __restrict__ cW3, const float* __restrict__ cb3,
    const _Float16* __restrict__ fW1, const float* __restrict__ fb1,
    const _Float16* __restrict__ fW2, const float* __restrict__ fb2,
    const _Float16* __restrict__ fW3, const float* __restrict__ fb3,
    float* __restrict__ xout, float invN) {
  __shared__ _Float16 H[8 * 16 * LDH];
  const int tid = threadIdx.x, wave = tid >> 5, lane = tid & 31;
  _Float16* Hw = H + wave * (16 * LDH);
  const int node0 = blockIdx.x * 128 + wave * 16;
  const v8f vzero = {0.f, 0.f, 0.f, 0.f, 0.f, 0.f, 0.f, 0.f};

  // ---- cur chain -----------------------------------------------------------
  v8f acc[8];
#pragma unroll
  for (int i = 0; i < 8; ++i) acc[i] = vzero;
  gemm_gA<2, 8>(x, 64, node0, cW1, acc, lane);
  store_act_lds<8, true>(acc, cb1, Hw, lane);
#pragma unroll
  for (int i = 0; i < 8; ++i) acc[i] = vzero;
  gemm_lA<4, 8>(Hw, cW2, acc, lane);
  store_act_lds<8, true>(acc, cb2, Hw, lane);
  v8f acc3[4];
#pragma unroll
  for (int i = 0; i < 4; ++i) acc3[i] = vzero;
  gemm_lA<4, 4>(Hw, cW3, acc3, lane);

  // ---- avg = (cur + s - inn) * invN -> LDS (f16, cols 0..63) ---------------
  const int nl = lane & 15, mb = (lane >> 4) << 3;
#pragma unroll
  for (int nt = 0; nt < 4; ++nt) {
    const float bv = cb3[nt * 16 + nl];
    const float sv = s[nt * 16 + nl];
#pragma unroll
    for (int r = 0; r < 8; ++r) {
      const int row = node0 + mb + r;
      const float cv = acc3[nt][r] + bv;
      const float av = (cv + sv - inn[(size_t)row * 64 + nt * 16 + nl]) * invN;
      Hw[(mb + r) * LDH + nt * 16 + nl] = (_Float16)av;
    }
  }

  // ---- fin chain (reads avg from LDS; DS is in-order per wave) -------------
#pragma unroll
  for (int i = 0; i < 8; ++i) acc[i] = vzero;
  gemm_lA<2, 8>(Hw, fW1, acc, lane);                   // 64 -> 128
  store_act_lds<8, true>(acc, fb1, Hw, lane);
#pragma unroll
  for (int i = 0; i < 8; ++i) acc[i] = vzero;
  gemm_lA<4, 8>(Hw, fW2, acc, lane);                   // 128 -> 128
  store_act_lds<8, true>(acc, fb2, Hw, lane);
#pragma unroll
  for (int i = 0; i < 4; ++i) acc3[i] = vzero;
  gemm_lA<4, 4>(Hw, fW3, acc3, lane);                  // 128 -> 64
#pragma unroll
  for (int nt = 0; nt < 4; ++nt) {
    const float bv = fb3[nt * 16 + nl];
#pragma unroll
    for (int r = 0; r < 8; ++r)
      xout[(size_t)(node0 + mb + r) * 64 + nt * 16 + nl] = acc3[nt][r] + bv;
  }
}

// ---------------------------------------------------------------------------
// Head: concat(x, z) -> 96->128->128->1 -> sigmoid.
// ---------------------------------------------------------------------------
__global__ void __launch_bounds__(256) k_head(
    const float* __restrict__ x, const float* __restrict__ z,
    const _Float16* __restrict__ hW1, const float* __restrict__ hb1,
    const _Float16* __restrict__ hW2, const float* __restrict__ hb2,
    const float* __restrict__ hW3, const float* __restrict__ hb3,
    float* __restrict__ out) {
  __shared__ _Float16 H[8 * 16 * LDH];
  const int tid = threadIdx.x, wave = tid >> 5, lane = tid & 31;
  _Float16* Hw = H + wave * (16 * LDH);
  const int node0 = blockIdx.x * 128 + wave * 16;
  const v8f vzero = {0.f, 0.f, 0.f, 0.f, 0.f, 0.f, 0.f, 0.f};

  v8f acc[8];
#pragma unroll
  for (int i = 0; i < 8; ++i) acc[i] = vzero;
  // K = 96: two 32-chunks from x, one broadcast chunk from z
#pragma unroll
  for (int kc = 0; kc < 2; ++kc) {
    const v16h a = load_A_g(x, 64, node0, kc * 32, lane);
    wmma_row<8>(a, hW1, kc, acc, lane);
  }
  {
    const v16h a = load_A_zbc(z, lane);
    wmma_row<8>(a, hW1, 2, acc, lane);
  }
  store_act_lds<8, true>(acc, hb1, Hw, lane);
#pragma unroll
  for (int i = 0; i < 8; ++i) acc[i] = vzero;
  gemm_lA<4, 8>(Hw, hW2, acc, lane);
  store_act_lds<8, true>(acc, hb2, Hw, lane);

  if (lane < 16) {   // 128 -> 1: plain dot product per node row
    float t = hb3[0];
#pragma unroll 16
    for (int c = 0; c < 128; ++c)
      t += (float)Hw[lane * LDH + c] * hW3[c];
    out[node0 + lane] = 1.0f / (1.0f + __expf(-t));
  }
}

// ---------------------------------------------------------------------------
extern "C" void kernel_launch(void* const* d_in, const int* in_sizes, int n_in,
                              void* d_out, int out_size, void* d_ws, size_t ws_size,
                              hipStream_t stream) {
  const float* vertex = (const float*)d_in[0];
  const float* z      = (const float*)d_in[1];
  // cur_params
  const float* cW1f = (const float*)d_in[2];  const float* cb1 = (const float*)d_in[3];
  const float* cW2f = (const float*)d_in[4];  const float* cb2 = (const float*)d_in[5];
  const float* cW3f = (const float*)d_in[6];  const float* cb3 = (const float*)d_in[7];
  // in_params
  const float* iW1f = (const float*)d_in[8];  const float* ib1 = (const float*)d_in[9];
  const float* iW2f = (const float*)d_in[10]; const float* ib2 = (const float*)d_in[11];
  const float* iW3f = (const float*)d_in[12]; const float* ib3 = (const float*)d_in[13];
  // fin_params
  const float* fW1f = (const float*)d_in[14]; const float* fb1 = (const float*)d_in[15];
  const float* fW2f = (const float*)d_in[16]; const float* fb2 = (const float*)d_in[17];
  const float* fW3f = (const float*)d_in[18]; const float* fb3 = (const float*)d_in[19];
  // head_params
  const float* hW1f = (const float*)d_in[20]; const float* hb1 = (const float*)d_in[21];
  const float* hW2f = (const float*)d_in[22]; const float* hb2 = (const float*)d_in[23];
  const float* hW3  = (const float*)d_in[24]; const float* hb3 = (const float*)d_in[25];

  const int n_nodes = in_sizes[0] / 64;
  const int nblocks = n_nodes / 128;

  // ---- workspace layout (half offsets for packed weights) ------------------
  _Float16* wh = (_Float16*)d_ws;
  _Float16* iW1 = wh;           _Float16* iW2 = wh + 8192;   _Float16* iW3 = wh + 24576;
  _Float16* cW1 = wh + 32768;   _Float16* cW2 = wh + 40960;  _Float16* cW3 = wh + 57344;
  _Float16* fW1 = wh + 65536;   _Float16* fW2 = wh + 73728;  _Float16* fW3 = wh + 90112;
  _Float16* hW1 = wh + 98304;   _Float16* hW2 = wh + 110592; // end: 126976 halves
  char* base = (char*)d_ws;
  float* s       = (float*)(base + 253952);                   // 64 floats
  float* partial = (float*)(base + 254208);                   // nblocks*64 floats
  float* inn     = (float*)(base + 778496);                   // n_nodes*64 floats
  float* xbuf    = (float*)(base + 778496 + (size_t)n_nodes * 64 * sizeof(float));

  // ---- pack weights to f16 fragment order (cheap; L2 keeps them hot) -------
#define PACK(src, dst, K, N) \
  k_pack<<<((K) * (N) + 255) / 256, 256, 0, stream>>>((src), (dst), (K), (N))
  PACK(iW1f, iW1, 64, 128);  PACK(iW2f, iW2, 128, 128); PACK(iW3f, iW3, 128, 64);
  PACK(cW1f, cW1, 64, 128);  PACK(cW2f, cW2, 128, 128); PACK(cW3f, cW3, 128, 64);
  PACK(fW1f, fW1, 64, 128);  PACK(fW2f, fW2, 128, 128); PACK(fW3f, fW3, 128, 64);
  PACK(hW1f, hW1, 96, 128);  PACK(hW2f, hW2, 128, 128);
#undef PACK

  const dim3 grid(nblocks), blk(256);
  const float invN = 1.0f / (float)n_nodes;

  const float* xin = vertex;
  for (int step = 0; step < 3; ++step) {
    k_msg_in<<<grid, blk, 0, stream>>>(xin, iW1, ib1, iW2, ib2, iW3, ib3, inn, partial);
    k_reduce<<<1, 64, 0, stream>>>(partial, s, nblocks);
    k_msg_fin<<<grid, blk, 0, stream>>>(xin, inn, s,
                                        cW1, cb1, cW2, cb2, cW3, cb3,
                                        fW1, fb1, fW2, fb2, fW3, fb3,
                                        xbuf, invN);
    xin = xbuf;
  }
  k_head<<<grid, blk, 0, stream>>>(xin, z, hW1, hb1, hW2, hb2, hW3, hb3, (float*)d_out);
}
